// SimpleSOTAAttention_77610059038934
// MI455X (gfx1250) — matrix operand order repaired and checked by reading, not compiled, not run on hardware
//
#include <hip/hip_runtime.h>
#include <hip/hip_bf16.h>

// ---------------- types ----------------
typedef _Float16 half_t;
typedef __attribute__((ext_vector_type(16))) _Float16 v16h;
typedef __attribute__((ext_vector_type(8)))  _Float16 v8h;
typedef __attribute__((ext_vector_type(4)))  _Float16 v4h;
typedef __attribute__((ext_vector_type(8)))  float    v8f;
typedef __attribute__((ext_vector_type(4)))  float    v4f;

#define DEVFN __device__ __forceinline__

// async global->LDS path (gfx1250 GLOBAL_LOAD_ASYNC_TO_LDS_*). Probe round 2
// showed the builtin exists with param0 = v4i in addrspace(1).
#if defined(__gfx1250__) && \
    __has_builtin(__builtin_amdgcn_global_load_async_to_lds_b128) && \
    __has_builtin(__builtin_amdgcn_s_wait_asynccnt)
#define USE_ASYNC_LDS 1
typedef int v4i_t __attribute__((ext_vector_type(4)));
typedef __attribute__((address_space(1))) v4i_t* gv4i_p;
typedef __attribute__((address_space(3))) v4i_t* lv4i_p;
#else
#define USE_ASYNC_LDS 0
#endif

// ---------------- problem constants ----------------
constexpr int BATCH = 4;
constexpr int SEQ   = 2048;
constexpr int HID   = 1024;
constexpr int NHEAD = 16;
constexpr int HDIM  = 64;
constexpr int MROWS = BATCH * SEQ;     // 8192

// ---------------- WMMA helpers ----------------
DEVFN v8f wmma32(v16h a, v16h b, v8f c) {
    // D = A(16x32 f16) * B(32x16 f16) + C(16x16 f32)
    return __builtin_amdgcn_wmma_f32_16x16x32_f16(
        false, a, false, b, (short)0, c, false, false);
}

// A-matrix fragment (16x32, f16). ISA layout: lanes 0-15 hold row M=lane,
// K = {0..7, 16..23}; lanes 16-31 hold row M=lane-16, K = {8..15, 24..31}.
DEVFN v16h load_a_frag(const half_t* base, int ld) {
    const int lane = threadIdx.x & 31;
    const int m  = lane & 15;
    const int kb = (lane >> 4) << 3;   // 0 or 8
    const half_t* p = base + m * ld + kb;
    v8h lo = *(const v8h*)(p);
    v8h hi = *(const v8h*)(p + 16);
    return __builtin_shufflevector(lo, hi, 0,1,2,3,4,5,6,7,8,9,10,11,12,13,14,15);
}

// B-matrix fragment (32x16, f16). Lanes 0-15 hold column N=lane, K = {0..7, 16..23};
// lanes 16-31 hold column N=lane-16, K = {8..15, 24..31}. Tile stored [N][K] row-major.
DEVFN v16h load_b_frag(const half_t* base, int ld) {
    const int lane = threadIdx.x & 31;
    const int n  = lane & 15;
    const int kb = (lane >> 4) << 3;   // 0 or 8
    const half_t* p = base + n * ld + kb;
    v8h lo = *(const v8h*)(p);
    v8h hi = *(const v8h*)(p + 16);
    return __builtin_shufflevector(lo, hi, 0,1,2,3,4,5,6,7,8,9,10,11,12,13,14,15);
}

// ---------------- f32 -> f16 convert ----------------
__global__ __launch_bounds__(256)
void cvt_f32_f16_kernel(const float* __restrict__ in, half_t* __restrict__ out, int n4) {
    int i = blockIdx.x * 256 + threadIdx.x;
    if (i < n4) {
        v4f v = ((const v4f*)in)[i];
        v4h o;
        o[0] = (half_t)v[0]; o[1] = (half_t)v[1];
        o[2] = (half_t)v[2]; o[3] = (half_t)v[3];
        ((v4h*)out)[i] = o;
    }
}

// ---------------- GEMM: C[M,N] = A[M,K] * W[N,K]^T ----------------
// MODE 0: f16 output scattered to [B, NH, S, HD]   (QKV projections)
// MODE 2: f32 output row-major [M, HID]            (final projection)
template<int MODE>
__global__ __launch_bounds__(256)
void gemm_kernel(const half_t* __restrict__ A, const half_t* __restrict__ W,
                 void* __restrict__ Cout) {
    constexpr int LDT = 40;                 // LDS row stride (halves), 16B aligned
    __shared__ __align__(16) half_t As[128 * LDT];
    __shared__ __align__(16) half_t Bs[128 * LDT];

    const int tid  = threadIdx.x;
    const int wid  = tid >> 5;
    const int lane = tid & 31;
    const int wm   = wid & 3;               // wave row   (4 x 32 rows)
    const int wn   = wid >> 2;              // wave col   (2 x 64 cols)
    const int bm   = blockIdx.x * 128;
    const int bn   = blockIdx.y * 128;

    v8f acc[2][4];
    v8f zv = {};
#pragma unroll
    for (int i = 0; i < 2; ++i)
#pragma unroll
        for (int j = 0; j < 4; ++j) acc[i][j] = zv;

    for (int kk = 0; kk < HID; kk += 32) {
        __syncthreads();
        // stage A tile 128x32 and W tile 128x32 (both K-contiguous)
#if USE_ASYNC_LDS
#pragma unroll
        for (int i = tid; i < 512; i += 256) {
            const int r = i >> 2;
            const int c = (i & 3) * 8;
            __builtin_amdgcn_global_load_async_to_lds_b128(
                (gv4i_p)(A + (size_t)(bm + r) * HID + kk + c),
                (lv4i_p)(As + r * LDT + c), 0, 0);
            __builtin_amdgcn_global_load_async_to_lds_b128(
                (gv4i_p)(W + (size_t)(bn + r) * HID + kk + c),
                (lv4i_p)(Bs + r * LDT + c), 0, 0);
        }
        __builtin_amdgcn_s_wait_asynccnt(0);
#else
#pragma unroll
        for (int i = tid; i < 512; i += 256) {
            const int r = i >> 2;
            const int c = (i & 3) * 8;
            *(v8h*)(As + r * LDT + c) =
                *(const v8h*)(A + (size_t)(bm + r) * HID + kk + c);
            *(v8h*)(Bs + r * LDT + c) =
                *(const v8h*)(W + (size_t)(bn + r) * HID + kk + c);
        }
#endif
        __syncthreads();

        // prefetch next K-tile into cache while this one computes
        if (kk + 32 < HID) {
            const half_t* pf = (tid < 128)
                ? (A + (size_t)(bm + tid) * HID + kk + 32)
                : (W + (size_t)(bn + tid - 128) * HID + kk + 32);
            __builtin_prefetch(pf, 0, 1);
        }

        v16h af[2], bf[4];
#pragma unroll
        for (int i = 0; i < 2; ++i)
            af[i] = load_a_frag(As + (wm * 32 + i * 16) * LDT, LDT);
#pragma unroll
        for (int j = 0; j < 4; ++j)
            bf[j] = load_b_frag(Bs + (wn * 64 + j * 16) * LDT, LDT);
#pragma unroll
        for (int i = 0; i < 2; ++i)
#pragma unroll
            for (int j = 0; j < 4; ++j)
                acc[i][j] = wmma32(af[i], bf[j], acc[i][j]);
    }

    // epilogue: C layout lane->(col = lane&15, rows r + 8*(lane>>4))
    const int colL = lane & 15;
    const int rowH = (lane >> 4) * 8;
#pragma unroll
    for (int i = 0; i < 2; ++i) {
#pragma unroll
        for (int j = 0; j < 4; ++j) {
            const int gn = bn + wn * 64 + j * 16 + colL;
#pragma unroll
            for (int r = 0; r < 8; ++r) {
                const int gm = bm + wm * 32 + i * 16 + rowH + r;
                const float v = acc[i][j][r];
                if (MODE == 0) {
                    const int b = gm >> 11;          // / SEQ
                    const int s = gm & (SEQ - 1);
                    const int h = gn >> 6;           // / HDIM
                    const int d = gn & (HDIM - 1);
                    ((half_t*)Cout)[(((size_t)b * NHEAD + h) * SEQ + s) * HDIM + d] =
                        (half_t)v;
                } else {
                    ((float*)Cout)[(size_t)gm * HID + gn] = v;
                }
            }
        }
    }
}

// ---------------- Flash attention ----------------
// grid.x = BATCH*NHEAD (64), grid.y = SEQ/64 (32), block = 128 threads (4 waves)
// Each wave owns 16 query rows. Online softmax over 32-key blocks; the running
// softmax denominator L is accumulated on the tensor core: L += P * ones.
__global__ __launch_bounds__(128)
void attn_kernel(const half_t* __restrict__ Qh, const half_t* __restrict__ Kh,
                 const half_t* __restrict__ Vh, const float* __restrict__ mask,
                 half_t* __restrict__ Ctx) {
    constexpr int LDV = 40;
    constexpr int LDP = 40;
    __shared__ __align__(16) half_t Vt[HDIM * LDV];       // V transposed: [d][k]
    __shared__ __align__(16) half_t Pw[4 * 16 * LDP];     // per-wave P tiles

    const int tid  = threadIdx.x;
    const int wid  = tid >> 5;
    const int lane = tid & 31;
    const int bh   = blockIdx.x;           // b*NHEAD + h
    const int b    = bh >> 4;
    const int h    = bh & (NHEAD - 1);
    const int q0   = blockIdx.y * 64 + wid * 16;

    const half_t* Qbase = Qh + ((size_t)bh * SEQ + q0) * HDIM;
    const half_t* Kbase = Kh + (size_t)bh * SEQ * HDIM;
    const half_t* Vbase = Vh + (size_t)bh * SEQ * HDIM;
    const float*  mrow  = mask + (size_t)b * SEQ;

    // Q fragments straight from global: [s][d] row-major == A-frag layout
    const v16h aQ0 = load_a_frag(Qbase,      HDIM);   // d 0..31
    const v16h aQ1 = load_a_frag(Qbase + 32, HDIM);   // d 32..63

    // ones B-fragment for tensor-core row sums
    v16h bOnes;
#pragma unroll
    for (int i = 0; i < 16; ++i) bOnes[i] = (half_t)1.0f;

    v8f oacc[4];
    v8f zv = {};
#pragma unroll
    for (int j = 0; j < 4; ++j) oacc[j] = zv;
    v8f lacc = zv;                         // running row-sum (replicated per col)
    float mrun[8];
#pragma unroll
    for (int r = 0; r < 8; ++r) mrun[r] = -1e30f;

    const float scale = 0.125f;            // HDIM^-0.5
    half_t* P = Pw + wid * 16 * LDP;
    const int colL = lane & 15;
    const int rowH = (lane >> 4) * 8;

    for (int kb = 0; kb < SEQ; kb += 32) {
        __syncthreads();
        // stage V block transposed: Vt[d][k] = V[kb+k][d]
        for (int i = tid; i < 32 * HDIM; i += 128) {
            const int k = i >> 6;
            const int d = i & (HDIM - 1);
            Vt[d * LDV + k] = Vbase[(size_t)(kb + k) * HDIM + d];
        }
        __syncthreads();

        // prefetch next K/V blocks (one cacheline per row)
        if (kb + 32 < SEQ) {
            const int pr = kb + 32 + lane;
            __builtin_prefetch(Kbase + (size_t)pr * HDIM, 0, 1);
            __builtin_prefetch(Vbase + (size_t)pr * HDIM, 0, 1);
        }

        // scores: S[16q, 32k] as two 16x16 tiles, contraction over d
        v8f sac[2];
#pragma unroll
        for (int t = 0; t < 2; ++t) {
            const half_t* kp = Kbase + (size_t)(kb + t * 16) * HDIM;
            v16h b0 = load_b_frag(kp,      HDIM);
            v16h b1 = load_b_frag(kp + 32, HDIM);
            v8f z = zv;
            z = wmma32(aQ0, b0, z);
            z = wmma32(aQ1, b1, z);
            sac[t] = z;
        }

        const float mk0 = mrow[kb + colL];
        const float mk1 = mrow[kb + 16 + colL];

        float alpha[8];
#pragma unroll
        for (int r = 0; r < 8; ++r) {
            float t0 = sac[0][r] * scale + mk0;
            float t1 = sac[1][r] * scale + mk1;
            float mx = fmaxf(t0, t1);
#pragma unroll
            for (int off = 1; off < 16; off <<= 1)
                mx = fmaxf(mx, __shfl_xor(mx, off, 32));
            const float mnew = fmaxf(mrun[r], mx);
            const float p0 = __expf(t0 - mnew);
            const float p1 = __expf(t1 - mnew);
            alpha[r] = __expf(mrun[r] - mnew);
            mrun[r] = mnew;
            // stash P (f16) into wave-private LDS in C layout
            P[(rowH + r) * LDP + colL]      = (half_t)p0;
            P[(rowH + r) * LDP + 16 + colL] = (half_t)p1;
        }
        // rescale running output and running denominator
#pragma unroll
        for (int j = 0; j < 4; ++j)
#pragma unroll
            for (int r = 0; r < 8; ++r) oacc[j][r] *= alpha[r];
#pragma unroll
        for (int r = 0; r < 8; ++r) lacc[r] *= alpha[r];

        // O += P[16q,32k] * V[32k,64d] ; L += P * ones (tensor-core row sum)
        const v16h aP = load_a_frag(P, LDP);
        lacc = wmma32(aP, bOnes, lacc);
#pragma unroll
        for (int j = 0; j < 4; ++j) {
            v16h bv = load_b_frag(Vt + j * 16 * LDV, LDV);
            oacc[j] = wmma32(aP, bv, oacc[j]);
        }
    }

    // normalize and write context [B, S, H] (f16 for final GEMM)
#pragma unroll
    for (int j = 0; j < 4; ++j) {
#pragma unroll
        for (int r = 0; r < 8; ++r) {
            const float v = oacc[j][r] / lacc[r];
            const int q = q0 + rowH + r;
            const int d = j * 16 + colL;
            Ctx[((size_t)b * SEQ + q) * HID + h * HDIM + d] = (half_t)v;
        }
    }
}

// ---------------- launcher ----------------
extern "C" void kernel_launch(void* const* d_in, const int* in_sizes, int n_in,
                              void* d_out, int out_size, void* d_ws, size_t ws_size,
                              hipStream_t stream) {
    const float* X    = (const float*)d_in[0];   // [B,S,H]
    const float* mask = (const float*)d_in[1];   // [B,S]
    const float* Wq   = (const float*)d_in[2];   // [H,H]
    const float* Wk   = (const float*)d_in[3];
    const float* Wv   = (const float*)d_in[4];
    const float* Wo   = (const float*)d_in[5];
    float* out = (float*)d_out;

    char* ws = (char*)d_ws;
    size_t off = 0;
    auto take = [&](size_t elems) { half_t* p = (half_t*)(ws + off); off += elems * sizeof(half_t); return p; };
    half_t* Xh   = take((size_t)MROWS * HID);
    half_t* Wqh  = take((size_t)HID * HID);
    half_t* Wkh  = take((size_t)HID * HID);
    half_t* Wvh  = take((size_t)HID * HID);
    half_t* Woh  = take((size_t)HID * HID);
    half_t* Qh   = take((size_t)MROWS * HID);
    half_t* Kh   = take((size_t)MROWS * HID);
    half_t* Vh   = take((size_t)MROWS * HID);
    half_t* Ctxh = take((size_t)MROWS * HID);

    // f32 -> f16 conversions
    {
        const int nx4 = MROWS * HID / 4;
        cvt_f32_f16_kernel<<<nx4 / 256, 256, 0, stream>>>(X, Xh, nx4);
        const int nw4 = HID * HID / 4;
        cvt_f32_f16_kernel<<<nw4 / 256, 256, 0, stream>>>(Wq, Wqh, nw4);
        cvt_f32_f16_kernel<<<nw4 / 256, 256, 0, stream>>>(Wk, Wkh, nw4);
        cvt_f32_f16_kernel<<<nw4 / 256, 256, 0, stream>>>(Wv, Wvh, nw4);
        cvt_f32_f16_kernel<<<nw4 / 256, 256, 0, stream>>>(Wo, Woh, nw4);
    }

    // QKV projections (f16 out, head-split layout)
    dim3 gg(MROWS / 128, HID / 128);
    gemm_kernel<0><<<gg, 256, 0, stream>>>(Xh, Wqh, Qh);
    gemm_kernel<0><<<gg, 256, 0, stream>>>(Xh, Wkh, Kh);
    gemm_kernel<0><<<gg, 256, 0, stream>>>(Xh, Wvh, Vh);

    // flash attention -> context (f16, [B,S,H])
    attn_kernel<<<dim3(BATCH * NHEAD, SEQ / 64), 128, 0, stream>>>(Qh, Kh, Vh, mask, Ctxh);

    // output projection (f32 out)
    gemm_kernel<2><<<gg, 256, 0, stream>>>(Ctxh, Woh, out);
}
